// LSTMLanguageModel_27968827032092
// MI455X (gfx1250) — matrix operand order repaired and checked by reading, not compiled, not run on hardware
//
#include <hip/hip_runtime.h>
#include <hip/hip_bf16.h>

typedef unsigned short u16;
typedef unsigned int   u32;
typedef __attribute__((ext_vector_type(16))) __bf16 v16bf;
typedef __attribute__((ext_vector_type(8)))  float  v8f;
typedef __attribute__((ext_vector_type(8)))  u32    v8u;
typedef __attribute__((__vector_size__(4 * sizeof(int)))) int i4v;

#define BV 128    // vocab
#define DD 1024   // embed dim
#define HH 1024   // hidden
#define LL 2
#define BB 64
#define TT 512
#define G4 4096   // 4*H

// LDS row strides (ushorts) for 64-wide K tiles (+8 pad -> 144B rows, 16B aligned)
#define KC 64
#define ASTR 72
#define BSTR 72

// -------- CDNA5 async global->LDS path (guarded; sync fallback keeps compiling)
#if defined(__AMDGCN__) && __has_builtin(__builtin_amdgcn_global_load_async_to_lds_b128)
#define USE_ASYNC_LDS 1
#else
#define USE_ASYNC_LDS 0
#endif

#if USE_ASYNC_LDS
#if __has_builtin(__builtin_amdgcn_s_wait_asynccnt)
#define WAIT_ASYNC(n) __builtin_amdgcn_s_wait_asynccnt(n)
#else
#define WAIT_ASYNC(n) asm volatile("s_wait_asynccnt " #n ::: "memory")
#endif
#else
#define WAIT_ASYNC(n) ((void)0)
#endif

__device__ __forceinline__ void cp16_lds(u16* l, const u16* g) {
#if USE_ASYNC_LDS
  __builtin_amdgcn_global_load_async_to_lds_b128(
      (__attribute__((address_space(1))) i4v*)g,
      (__attribute__((address_space(3))) i4v*)l, 0, 0);
#else
  *(uint4*)l = *(const uint4*)g;
#endif
}

__device__ __forceinline__ u16 f2bf(float f) {
  u32 u = __builtin_bit_cast(u32, f);
  u32 r = (u + 0x7FFFu + ((u >> 16) & 1u)) >> 16;   // round-to-nearest-even
  return (u16)r;
}
__device__ __forceinline__ float bf2f(u16 v) {
  return __builtin_bit_cast(float, (u32)v << 16);
}
__device__ __forceinline__ float sigmf(float x) { return 1.f / (1.f + __expf(-x)); }

// A fragment: 16x32 bf16 at a 32-wide K window of an LDS tile (row-major).
// Layout: lanes 0-15 hold M rows; VGPR j: K=2j,2j+1 (+16 upper half),
// matching frag_b's contiguous K order.
__device__ __forceinline__ v16bf frag_a(const u16* row0, int stride, int lane) {
  const u16* row = row0 + (lane & 15) * stride;
  const int half8 = ((lane >> 4) & 1) * 16;
  v8u p;
#pragma unroll
  for (int j = 0; j < 8; ++j) {
    int k = half8 + 2 * j;
    p[j] = (u32)row[k] | ((u32)row[k + 1] << 16);
  }
  return __builtin_bit_cast(v16bf, p);
}

// B fragment: 32x16 bf16, tile stored [n][k]. Lane = N col; lanes 0-15 K=0..15,
// lanes 16-31 K=16..31 -> 16 contiguous ushorts (two ds_load_b128).
__device__ __forceinline__ v16bf frag_b(const u16* brow0, int stride, int lane) {
  const u16* src = brow0 + (lane & 15) * stride + ((lane >> 4) & 1) * 16;
  v8u p;
#pragma unroll
  for (int j = 0; j < 8; ++j)
    p[j] = (u32)src[2 * j] | ((u32)src[2 * j + 1] << 16);
  return __builtin_bit_cast(v16bf, p);
}

// ---------------------------------------------------------------- converts
__global__ __launch_bounds__(256)
void cvt_f32_bf16(const float* __restrict__ src, u16* __restrict__ dst, int n) {
  int i = (blockIdx.x * 256 + threadIdx.x) * 4;
  if (i + 3 < n) {
    float4 v = *(const float4*)(src + i);
    u32 lo = (u32)f2bf(v.x) | ((u32)f2bf(v.y) << 16);
    u32 hi = (u32)f2bf(v.z) | ((u32)f2bf(v.w) << 16);
    *(uint2*)(dst + i) = make_uint2(lo, hi);
  }
}

// --------------------------------------------- x-projection GEMM (parallel)
// G[m, n] = A_row(m) . W[n, :]   (bf16 in, bf16 out, f32 accumulate)
// A_row(m) = Abase + tok(m)*K when xtok gather enabled (layer 1: embed table),
// else Abase + m*K (layer 2: H1). M = T*B, N = 4H, K = 1024.
// Block 128(M) x 128(N), 8 waves in 4x2, each wave 2x4 16x16 tiles, K chunk 64.
__global__ __launch_bounds__(256)
void gemm_xproj(const u16* __restrict__ Abase, const int* __restrict__ xtok,
                int K, const u16* __restrict__ W, u16* __restrict__ G) {
  __shared__ u16 As[128 * ASTR];
  __shared__ u16 Bs[128 * BSTR];
  const int tid = threadIdx.x, lane = tid & 31, wave = tid >> 5;
  const int wm0 = (wave >> 1) * 32, wn0 = (wave & 1) * 64;
  const int m0 = blockIdx.x * 128, n0 = blockIdx.y * 128;

  v8f acc[2][4];
#pragma unroll
  for (int i = 0; i < 2; ++i)
#pragma unroll
    for (int j = 0; j < 4; ++j)
#pragma unroll
      for (int e = 0; e < 8; ++e) acc[i][j][e] = 0.f;

  // A row base pointer for this thread's staging row (2 threads per row)
  const int srow = tid >> 1;
  const int scs = (tid & 1) * 32;
  const u16* arow;
  if (xtok) {
    int gm = m0 + srow;                          // m = t*B + b
    int tok = xtok[(gm & 63) * TT + (gm >> 6)];  // x is [B,T]
    arow = Abase + (size_t)tok * K;
  } else {
    arow = Abase + (size_t)(m0 + srow) * K;
  }
  const u16* brow = W + (size_t)(n0 + srow) * K;

  for (int k0 = 0; k0 < K; k0 += KC) {
    if (k0 + KC < K) {
      __builtin_prefetch(arow + k0 + KC + scs, 0, 1);
      __builtin_prefetch(brow + k0 + KC + scs, 0, 1);
    }
#pragma unroll
    for (int q = 0; q < 4; ++q) {
      uint4 va = ((const uint4*)(arow + k0 + scs))[q];
      *(uint4*)(As + srow * ASTR + scs + q * 8) = va;
    }
#pragma unroll
    for (int q = 0; q < 4; ++q) {
      uint4 vb = ((const uint4*)(brow + k0 + scs))[q];
      *(uint4*)(Bs + srow * BSTR + scs + q * 8) = vb;
    }
    __syncthreads();
#pragma unroll
    for (int ks = 0; ks < 2; ++ks) {
      v16bf bf[4];
#pragma unroll
      for (int nt = 0; nt < 4; ++nt)
        bf[nt] = frag_b(Bs + (wn0 + nt * 16) * BSTR + ks * 32, BSTR, lane);
#pragma unroll
      for (int mt = 0; mt < 2; ++mt) {
        v16bf af = frag_a(As + (wm0 + mt * 16) * ASTR + ks * 32, ASTR, lane);
#pragma unroll
        for (int nt = 0; nt < 4; ++nt)
          acc[mt][nt] = __builtin_amdgcn_wmma_f32_16x16x32_bf16(
              false, af, false, bf[nt], (short)0, acc[mt][nt], false, false);
      }
    }
    __syncthreads();
  }

  const int half = lane >> 4, nlo = lane & 15;
#pragma unroll
  for (int mt = 0; mt < 2; ++mt)
#pragma unroll
    for (int nt = 0; nt < 4; ++nt) {
      int n = n0 + wn0 + nt * 16 + nlo;
#pragma unroll
      for (int r = 0; r < 8; ++r) {
        int m = m0 + wm0 + mt * 16 + half * 8 + r;
        G[(size_t)m * G4 + n] = f2bf(acc[mt][nt][r]);
      }
    }
}

// ------------------------------------------------------- fused LSTM step
// gates[64,4H] = Gt (precomputed x-proj) + Hprev@Wh^T + b ; cell update for
// 32 h-cols/block. 8 waves: gate = w>>1, 16-col half = w&1; 4 M-tiles (64 batch).
// Double-buffered LDS; async global->LDS copies overlap with WMMA of the
// previous K chunk (wave pays only s_wait_asynccnt <= in-flight batch).
__global__ __launch_bounds__(256)
void lstm_step_wmma(const u16* __restrict__ Gt,     // [B, 4H] bf16 x-proj slice
                    const u16* __restrict__ Hprev,  // [B, H] bf16
                    const u16* __restrict__ Whl,    // [4H, H] bf16
                    const float* __restrict__ bias, // [4H]
                    float* __restrict__ Cst,        // [B, H] f32 state (in/out)
                    u16* __restrict__ Hout) {       // [B, H] bf16 h_t
  // carve: buf0 A(9216) | buf0 B(18432) | buf1 A(9216) | buf1 B(18432)
  // Gs (32768 f32 gate spill) aliases buf1 region (used only after K loop).
  __shared__ __align__(16) char smem[60416];
  u16* AsBuf0 = (u16*)smem;
  u16* BsBuf0 = (u16*)(smem + 9216);
  u16* AsBuf1 = (u16*)(smem + 27648);
  u16* BsBuf1 = (u16*)(smem + 36864);
  float(*Gs)[64][32] = (float(*)[64][32])(smem + 27648);

  const int tid = threadIdx.x;
  const int lane = tid & 31;
  const int wave = tid >> 5;
  const int gate = wave >> 1;
  const int nsub = wave & 1;
  const int n0 = blockIdx.x * 32;  // h-column base

  v8f acc[4];
#pragma unroll
  for (int i = 0; i < 4; ++i)
#pragma unroll
    for (int e = 0; e < 8; ++e) acc[i][e] = 0.f;

  // staging coords: A = 64 rows x 64 k (2 x b128/thread); B = 4x32 n x 64 k (4/thread)
  const int arow = tid >> 2, acs = (tid & 3) * 16;
  const int bg = tid >> 6, bn = (tid & 63) >> 1, bcs = (tid & 1) * 32;
  const u16* agp = Hprev + (size_t)arow * HH + acs;
  const u16* bgp = Whl + (size_t)(bg * HH + n0 + bn) * HH + bcs;

  auto stage = [&](u16* Asb, u16* Bsb, int k0) {
    u16* Ad = Asb + arow * ASTR + acs;
    const u16* Ag = agp + k0;
    cp16_lds(Ad, Ag);
    cp16_lds(Ad + 8, Ag + 8);
    u16* Bd = Bsb + (bg * 32 + bn) * BSTR + bcs;
    const u16* Bg = bgp + k0;
#pragma unroll
    for (int q = 0; q < 4; ++q) cp16_lds(Bd + q * 8, Bg + q * 8);
  };

  auto compute = [&](const u16* Asb, const u16* Bsb) {
#pragma unroll
    for (int ks = 0; ks < 2; ++ks) {
      v16bf bfrag = frag_b(Bsb + (gate * 32 + nsub * 16) * BSTR + ks * 32, BSTR, lane);
#pragma unroll
      for (int mt = 0; mt < 4; ++mt) {
        v16bf afrag = frag_a(Asb + mt * 16 * ASTR + ks * 32, ASTR, lane);
        acc[mt] = __builtin_amdgcn_wmma_f32_16x16x32_bf16(
            false, afrag, false, bfrag, (short)0, acc[mt], false, false);
      }
    }
  };

  // K = HH = 16 chunks of 64; software pipeline with 1 chunk in flight
  stage(AsBuf0, BsBuf0, 0);
  for (int ci = 0; ci < 15; ++ci) {
    if (ci & 1) stage(AsBuf0, BsBuf0, (ci + 1) * KC);
    else        stage(AsBuf1, BsBuf1, (ci + 1) * KC);
    WAIT_ASYNC(6);   // 6 async b128 in flight for next chunk -> current done
    __syncthreads();
    if (ci & 1) compute(AsBuf1, BsBuf1);
    else        compute(AsBuf0, BsBuf0);
    __syncthreads();
  }
  WAIT_ASYNC(0);
  __syncthreads();
  compute(AsBuf1, BsBuf1);  // chunk 15 -> buf1
  __syncthreads();          // all reads of buf1 done before Gs alias writes

  // spill gates (+ x-projection + bias) to LDS for the cross-gate cell update
  {
    int half = lane >> 4;
    int nl = nsub * 16 + (lane & 15);
    float bval = bias[gate * HH + n0 + nl];
    const u16* gcol = Gt + gate * HH + n0 + nl;
#pragma unroll
    for (int mt = 0; mt < 4; ++mt)
#pragma unroll
      for (int r = 0; r < 8; ++r) {
        int m = mt * 16 + half * 8 + r;
        Gs[gate][m][nl] = acc[mt][r] + bval + bf2f(gcol[(size_t)m * G4]);
      }
  }
  __syncthreads();

  // cell update: 64 batch x 32 cols
  for (int e = tid; e < 64 * 32; e += 256) {
    int m = e >> 5, col = e & 31;
    int hidx = n0 + col;
    float iv = Gs[0][m][col], fv = Gs[1][m][col];
    float gv = Gs[2][m][col], ov = Gs[3][m][col];
    float cold = Cst[m * HH + hidx];
    float cn = sigmf(fv) * cold + sigmf(iv) * tanhf(gv);
    Cst[m * HH + hidx] = cn;
    Hout[m * HH + hidx] = f2bf(sigmf(ov) * tanhf(cn));
  }
}

// --------------------------------------------------------- logits GEMM
// out[b,t,v] = H2[t*B+b,:] @ Wout[v,:] + bout[v]. M=T*B, N=V=128, K=H.
__global__ __launch_bounds__(256)
void logits_wmma(const u16* __restrict__ H2, const u16* __restrict__ Wo,
                 const float* __restrict__ bout, float* __restrict__ out) {
  __shared__ u16 As[128 * ASTR];
  __shared__ u16 Bs[128 * BSTR];
  const int tid = threadIdx.x, lane = tid & 31, wave = tid >> 5;
  const int wm0 = (wave >> 1) * 32, wn0 = (wave & 1) * 64;
  const int m0 = blockIdx.x * 128;

  v8f acc[2][4];
#pragma unroll
  for (int i = 0; i < 2; ++i)
#pragma unroll
    for (int j = 0; j < 4; ++j)
#pragma unroll
      for (int e = 0; e < 8; ++e) acc[i][j][e] = 0.f;

  const int srow = tid >> 1, scs = (tid & 1) * 32;
  const u16* arow = H2 + (size_t)(m0 + srow) * HH;
  const u16* brow = Wo + (size_t)srow * HH;

  for (int k0 = 0; k0 < HH; k0 += KC) {
    if (k0 + KC < HH) {
      __builtin_prefetch(arow + k0 + KC + scs, 0, 1);
      __builtin_prefetch(brow + k0 + KC + scs, 0, 1);
    }
#pragma unroll
    for (int q = 0; q < 4; ++q)
      *(uint4*)(As + srow * ASTR + scs + q * 8) =
          ((const uint4*)(arow + k0 + scs))[q];
#pragma unroll
    for (int q = 0; q < 4; ++q)
      *(uint4*)(Bs + srow * BSTR + scs + q * 8) =
          ((const uint4*)(brow + k0 + scs))[q];
    __syncthreads();
#pragma unroll
    for (int ks = 0; ks < 2; ++ks) {
      v16bf bf[4];
#pragma unroll
      for (int nt = 0; nt < 4; ++nt)
        bf[nt] = frag_b(Bs + (wn0 + nt * 16) * BSTR + ks * 32, BSTR, lane);
#pragma unroll
      for (int mt = 0; mt < 2; ++mt) {
        v16bf af = frag_a(As + (wm0 + mt * 16) * ASTR + ks * 32, ASTR, lane);
#pragma unroll
        for (int nt = 0; nt < 4; ++nt)
          acc[mt][nt] = __builtin_amdgcn_wmma_f32_16x16x32_bf16(
              false, af, false, bf[nt], (short)0, acc[mt][nt], false, false);
      }
    }
    __syncthreads();
  }

  const int half = lane >> 4, nlo = lane & 15;
#pragma unroll
  for (int mt = 0; mt < 2; ++mt)
#pragma unroll
    for (int nt = 0; nt < 4; ++nt) {
      int n = wn0 + nt * 16 + nlo;
      float bb = bout[n];
#pragma unroll
      for (int r = 0; r < 8; ++r) {
        int m = m0 + wm0 + mt * 16 + half * 8 + r;  // m = t*B + b
        int t = m >> 6, b = m & 63;
        out[((size_t)b * TT + t) * BV + n] = acc[mt][nt][r] + bb;
      }
    }
}

// ---------------------------------------------------------------- launch
static size_t align_up(size_t o) { return (o + 255) & ~(size_t)255; }

extern "C" void kernel_launch(void* const* d_in, const int* in_sizes, int n_in,
                              void* d_out, int out_size, void* d_ws, size_t ws_size,
                              hipStream_t stream) {
  const int*   x     = (const int*)d_in[0];     // [B,T]
  const float* embed = (const float*)d_in[1];   // [V,D]
  const float* Wx    = (const float*)d_in[2];   // [L,4H,D]
  const float* Wh    = (const float*)d_in[3];   // [L,4H,H]
  const float* bias  = (const float*)d_in[4];   // [L,4H]
  const float* Wout  = (const float*)d_in[5];   // [V,H]
  const float* bout  = (const float*)d_in[6];   // [V]
  float* out = (float*)d_out;

  char* ws = (char*)d_ws;
  size_t off = 0;
  auto take = [&](size_t bytes) -> char* {
    char* p = ws + off;
    off = align_up(off + bytes);
    return p;
  };
  u16* Ebf  = (u16*)take((size_t)BV * DD * 2);
  u16* Wxb  = (u16*)take((size_t)LL * G4 * DD * 2);
  u16* Whb  = (u16*)take((size_t)LL * G4 * HH * 2);
  u16* Wob  = (u16*)take((size_t)BV * HH * 2);
  u16* Gbuf = (u16*)take((size_t)TT * BB * G4 * 2);  // x-projection, reused per layer
  u16* H1   = (u16*)take((size_t)TT * BB * HH * 2);
  u16* H2   = (u16*)take((size_t)TT * BB * HH * 2);
  u16* h0   = (u16*)take((size_t)BB * HH * 2);
  float* cs = (float*)take((size_t)LL * BB * HH * 4);

  // bf16 converts
  cvt_f32_bf16<<<(BV * DD) / 1024, 256, 0, stream>>>(embed, Ebf, BV * DD);
  cvt_f32_bf16<<<(LL * G4 * DD) / 1024, 256, 0, stream>>>(Wx, Wxb, LL * G4 * DD);
  cvt_f32_bf16<<<(LL * G4 * HH) / 1024, 256, 0, stream>>>(Wh, Whb, LL * G4 * HH);
  cvt_f32_bf16<<<(BV * HH) / 1024, 256, 0, stream>>>(Wout, Wob, BV * HH);

  (void)hipMemsetAsync(h0, 0, (size_t)BB * HH * 2, stream);
  (void)hipMemsetAsync(cs, 0, (size_t)LL * BB * HH * 4, stream);

  // layer-1 x-projection (fused embedding gather): G = embed[x] @ Wx1^T
  gemm_xproj<<<dim3((TT * BB) / 128, G4 / 128), 256, 0, stream>>>(
      Ebf, x, DD, Wxb, Gbuf);

  // layer-1 recurrence
  for (int t = 0; t < TT; ++t) {
    const u16* Hp = t ? H1 + (size_t)(t - 1) * BB * HH : h0;
    lstm_step_wmma<<<HH / 32, 256, 0, stream>>>(
        Gbuf + (size_t)t * BB * G4, Hp, Whb, bias, cs,
        H1 + (size_t)t * BB * HH);
  }

  // layer-2 x-projection: G = H1 @ Wx2^T
  gemm_xproj<<<dim3((TT * BB) / 128, G4 / 128), 256, 0, stream>>>(
      H1, nullptr, HH, Wxb + (size_t)G4 * DD, Gbuf);

  // layer-2 recurrence
  for (int t = 0; t < TT; ++t) {
    const u16* Hp = t ? H2 + (size_t)(t - 1) * BB * HH : h0;
    lstm_step_wmma<<<HH / 32, 256, 0, stream>>>(
        Gbuf + (size_t)t * BB * G4, Hp, Whb + (size_t)G4 * HH, bias + G4,
        cs + (size_t)BB * HH, H2 + (size_t)t * BB * HH);
  }

  // logits
  logits_wmma<<<(TT * BB) / 128, 256, 0, stream>>>(H2, Wob, bout, out);
}
